// RefineBottleneck_90348932038889
// MI455X (gfx1250) — matrix operand order repaired and checked by reading, not compiled, not run on hardware
//
#include <hip/hip_runtime.h>
#include <math.h>

// ---------- types ----------
typedef __bf16 bf16;
typedef __attribute__((ext_vector_type(16))) __bf16 bf16x16;
typedef __attribute__((ext_vector_type(8)))  __bf16 bf16x8;
typedef __attribute__((ext_vector_type(4)))  __bf16 bf16x4;
typedef __attribute__((ext_vector_type(8)))  float  f32x8;
typedef __attribute__((ext_vector_type(4)))  int    v4i;

#define AS1 __attribute__((address_space(1)))
#define AS3 __attribute__((address_space(3)))

union Frag { bf16x16 v; uint4 q[2]; };

__device__ __forceinline__ f32x8 wmma_bf16(const bf16x16 a, const bf16x16 b, const f32x8 c) {
    return __builtin_amdgcn_wmma_f32_16x16x32_bf16(false, a, false, b, (short)0, c, false, false);
}
__device__ __forceinline__ f32x8 zero8() {
    f32x8 z = {0.f,0.f,0.f,0.f,0.f,0.f,0.f,0.f};
    return z;
}

// async global->LDS path (CDNA5); fall back to sync staging if builtin absent
#if defined(__has_builtin)
#if __has_builtin(__builtin_amdgcn_global_load_async_to_lds_b128)
#define ASYNC_LDS 1
#endif
#endif
#ifndef ASYNC_LDS
#define ASYNC_LDS 0
#endif

#if ASYNC_LDS
#if __has_builtin(__builtin_amdgcn_s_wait_asynccnt)
#define WAIT_ASYNC(n) __builtin_amdgcn_s_wait_asynccnt(n)
#else
#define WAIT_ASYNC(n) asm volatile("s_wait_asynccnt %0" ::"i"(n))
#endif
#else
#define WAIT_ASYNC(n)
#endif

// constants
#define BATCH 8
#define DIM   256
#define NTOK  64          // tokens per window (8x8)
#define NWIN  2048        // total windows = B * 16 * 16
#define MTOT  131072      // B*H*W tokens
#define HID   1024

// ---------- fp32 -> bf16 convert ----------
__global__ __launch_bounds__(256) void cvt_kernel(const float* __restrict__ in,
                                                  bf16* __restrict__ out, int n) {
    int i = blockIdx.x * 256 + threadIdx.x;
    if (i < n) out[i] = (bf16)in[i];
}

// ---------- LayerNorm (one wave per token, DIM=256) ----------
// remap=1: gather with roll(-4,-4) + window partition (token t in window order)
__global__ __launch_bounds__(256) void ln_kernel(const float* __restrict__ in,
                                                 const float* __restrict__ g,
                                                 const float* __restrict__ b,
                                                 float* __restrict__ outF,
                                                 bf16*  __restrict__ outB,
                                                 int remap) {
    int t    = blockIdx.x * 8 + (threadIdx.x >> 5);
    int lane = threadIdx.x & 31;
    size_t base;
    if (remap) {
        int wd = t >> 6, idx = t & 63;
        int bb = wd >> 8, wl = wd & 255, wh = wl >> 4, wwc = wl & 15;
        int hs = wh * 8 + (idx >> 3), ws = wwc * 8 + (idx & 7);
        int hsrc = (hs + 4) & 127, wsrc = (ws + 4) & 127;
        base = ((size_t)((bb << 14) + hsrc * 128 + wsrc)) * DIM;
    } else {
        base = (size_t)t * DIM;
    }
    const float4* p = (const float4*)(in + base + lane * 8);
    float4 a0 = p[0], a1 = p[1];
    float v[8] = {a0.x, a0.y, a0.z, a0.w, a1.x, a1.y, a1.z, a1.w};
    float s = 0.f;
#pragma unroll
    for (int j = 0; j < 8; ++j) s += v[j];
#pragma unroll
    for (int off = 1; off < 32; off <<= 1) s += __shfl_xor(s, off, 32);
    float mean = s * (1.f / 256.f);
    float vs = 0.f;
#pragma unroll
    for (int j = 0; j < 8; ++j) { float d = v[j] - mean; vs += d * d; }
#pragma unroll
    for (int off = 1; off < 32; off <<= 1) vs += __shfl_xor(vs, off, 32);
    float rstd = rsqrtf(vs * (1.f / 256.f) + 1e-5f);
    float y[8];
#pragma unroll
    for (int j = 0; j < 8; ++j) {
        int c = lane * 8 + j;
        y[j] = (v[j] - mean) * rstd * g[c] + b[c];
    }
    if (outF) {
        float4 o0 = {y[0], y[1], y[2], y[3]}, o1 = {y[4], y[5], y[6], y[7]};
        float4* q = (float4*)(outF + (size_t)t * DIM + lane * 8);
        q[0] = o0; q[1] = o1;
    }
    bf16x8 ob;
#pragma unroll
    for (int j = 0; j < 8; ++j) ob[j] = (bf16)y[j];
    *(bf16x8*)(outB + (size_t)t * DIM + lane * 8) = ob;
}

// ---------- generic bf16 WMMA GEMM:  C[M,N] = A[M,K] * W[N,K]^T  ----------
// block tile 128x256, 8 waves (2x4) of 64x64, double-buffered LDS, BK=32
// async global->LDS staging when available (6 async b128 per thread per stage)
// MODE 0: out bf16 (QKV, no bias)
// MODE 1: +bias +resid(xn) -> scatter (window-reverse + roll) into f32 xs
// MODE 2: +bias -> out bf16 (fc1)
// MODE 3: +bias +resid(xs) -> out f32 (fc2 -> d_out)
template <int MODE>
__global__ __launch_bounds__(256) void gemm_kernel(const bf16* __restrict__ A,
                                                   const bf16* __restrict__ Wt,
                                                   const float* __restrict__ bias,
                                                   const float* __restrict__ resid,
                                                   bf16*  __restrict__ outB,
                                                   float* __restrict__ outF,
                                                   int N, int K) {
    __shared__ bf16 sA[2][128 * 32];
    __shared__ bf16 sB[2][256 * 32];

    const int tid  = threadIdx.x;
    const int lane = tid & 31;
    const int hf   = lane >> 4;
    const int lr   = lane & 15;
    const int wave = tid >> 5;
    const int waveM = wave >> 2;   // 0..1
    const int waveN = wave & 3;    // 0..3
    const int blockRow = blockIdx.x * 128;
    const int blockCol = blockIdx.y * 256;

    f32x8 acc[4][4];
#pragma unroll
    for (int mi = 0; mi < 4; ++mi)
#pragma unroll
        for (int nj = 0; nj < 4; ++nj) acc[mi][nj] = zero8();

    auto stage = [&](int buf, int k0) {
#pragma unroll
        for (int p = 0; p < 2; ++p) {
            int row = (tid >> 2) + p * 64;
            int seg = tid & 3;
            const bf16* gp = A + (size_t)(blockRow + row) * K + k0 + seg * 8;
            bf16* lp = &sA[buf][row * 32 + seg * 8];
#if ASYNC_LDS
            __builtin_amdgcn_global_load_async_to_lds_b128(
                (AS1 v4i*)gp, (AS3 v4i*)lp, 0, 0);
#else
            *(uint4*)lp = *(const uint4*)gp;
#endif
        }
#pragma unroll
        for (int p = 0; p < 4; ++p) {
            int row = (tid >> 2) + p * 64;
            int seg = tid & 3;
            const bf16* gp = Wt + (size_t)(blockCol + row) * K + k0 + seg * 8;
            bf16* lp = &sB[buf][row * 32 + seg * 8];
#if ASYNC_LDS
            __builtin_amdgcn_global_load_async_to_lds_b128(
                (AS1 v4i*)gp, (AS3 v4i*)lp, 0, 0);
#else
            *(uint4*)lp = *(const uint4*)gp;
#endif
        }
    };

    const int nIter = K >> 5;
    stage(0, 0);
    for (int it = 0; it < nIter; ++it) {
        int buf = it & 1;
        if (it + 1 < nIter) {
            stage(buf ^ 1, (it + 1) << 5);
            WAIT_ASYNC(6);    // previous stage's 6 loads complete (in-order)
        } else {
            WAIT_ASYNC(0);
        }
        __syncthreads();

        Frag af[4], bfr[4];
#pragma unroll
        for (int mi = 0; mi < 4; ++mi) {
            const uint4* pa = (const uint4*)(&sA[buf][(waveM * 64 + mi * 16 + lr) * 32]);
            af[mi].q[0] = pa[hf];
            af[mi].q[1] = pa[2 + hf];
        }
#pragma unroll
        for (int nj = 0; nj < 4; ++nj) {
            const uint4* pb = (const uint4*)(&sB[buf][(waveN * 64 + nj * 16 + lr) * 32]);
            bfr[nj].q[0] = pb[hf];
            bfr[nj].q[1] = pb[2 + hf];
        }
#pragma unroll
        for (int mi = 0; mi < 4; ++mi)
#pragma unroll
            for (int nj = 0; nj < 4; ++nj)
                acc[mi][nj] = wmma_bf16(af[mi].v, bfr[nj].v, acc[mi][nj]);
        __syncthreads();   // reads done before this buffer is re-staged
    }

    // epilogue
#pragma unroll
    for (int mi = 0; mi < 4; ++mi)
#pragma unroll
        for (int nj = 0; nj < 4; ++nj)
#pragma unroll
            for (int r = 0; r < 8; ++r) {
                int row = blockRow + waveM * 64 + mi * 16 + r + 8 * hf;
                int col = blockCol + waveN * 64 + nj * 16 + lr;
                float v = acc[mi][nj][r];
                if (MODE == 0) {
                    outB[(size_t)row * N + col] = (bf16)v;
                } else if (MODE == 1) {
                    v += bias[col] + resid[(size_t)row * DIM + col];
                    int t = row, wd = t >> 6, idx = t & 63;
                    int bb = wd >> 8, wl = wd & 255, wh = wl >> 4, wwc = wl & 15;
                    int hs = wh * 8 + (idx >> 3), ws = wwc * 8 + (idx & 7);
                    int hd2 = (hs + 4) & 127, wd2 = (ws + 4) & 127;
                    size_t dst = ((size_t)((bb << 14) + hd2 * 128 + wd2)) * DIM + col;
                    outF[dst] = v;
                } else if (MODE == 2) {
                    v += bias[col];
                    outB[(size_t)row * N + col] = (bf16)v;
                } else {
                    v += bias[col] + resid[(size_t)row * DIM + col];
                    outF[(size_t)row * DIM + col] = v;
                }
            }
}

// ---------- shifted-window attention: 1 block = 1 window, 1 wave = 1 head ----------
__global__ __launch_bounds__(256) void attn_kernel(const bf16* __restrict__ qkv,
                                                   bf16* __restrict__ o) {
    extern __shared__ char smem[];
    unsigned char* regid = (unsigned char*)smem;                 // [64] (+pad to 128)
    bf16* P  = (bf16*)(smem + 128);                              // [8][64][64]
    bf16* VT = (bf16*)(smem + 128 + 8 * 64 * 64 * 2);            // [8][32][64]

    const int w    = blockIdx.x;
    const int tid  = threadIdx.x;
    const int wave = tid >> 5;
    const int lane = tid & 31;
    const int hf   = lane >> 4;
    const int lr   = lane & 15;
    const int wl   = w & 255, wh = wl >> 4, wwc = wl & 15;

    if (tid < 64) {
        int i = tid >> 3, j = tid & 7;
        int hs = wh * 8 + i, ws = wwc * 8 + j;
        int rr = hs < 120 ? 0 : (hs < 124 ? 1 : 2);
        int cc = ws < 120 ? 0 : (ws < 124 ? 1 : 2);
        regid[tid] = (unsigned char)(rr * 3 + cc);
    }
    // transpose V into LDS: VT[d][tok]
    bf16* vt = VT + wave * 32 * 64;
    {
        const bf16* vbase = qkv + (size_t)w * NTOK * 768 + 512 + wave * 32;
        for (int t = lane; t < 64; t += 32) {
            const bf16x8* vr = (const bf16x8*)(vbase + (size_t)t * 768);
            bf16x8 r0 = vr[0], r1 = vr[1], r2 = vr[2], r3 = vr[3];
#pragma unroll
            for (int d = 0; d < 8; ++d) {
                vt[d * 64 + t]        = r0[d];
                vt[(d + 8) * 64 + t]  = r1[d];
                vt[(d + 16) * 64 + t] = r2[d];
                vt[(d + 24) * 64 + t] = r3[d];
            }
        }
    }
    __syncthreads();

    // S = Q K^T  (K-dim = hd = 32 -> single WMMA per tile)
    f32x8 s[4][4];
#pragma unroll
    for (int mi = 0; mi < 4; ++mi)
#pragma unroll
        for (int nj = 0; nj < 4; ++nj) s[mi][nj] = zero8();

    const bf16* qbase = qkv + (size_t)w * NTOK * 768 + wave * 32;
    const bf16* kbase = qkv + (size_t)w * NTOK * 768 + 256 + wave * 32;
    Frag qa[4], kb[4];
#pragma unroll
    for (int mi = 0; mi < 4; ++mi) {
        const uint4* p = (const uint4*)(qbase + (size_t)(mi * 16 + lr) * 768);
        qa[mi].q[0] = p[hf];
        qa[mi].q[1] = p[2 + hf];
    }
#pragma unroll
    for (int nj = 0; nj < 4; ++nj) {
        const uint4* p = (const uint4*)(kbase + (size_t)(nj * 16 + lr) * 768);
        kb[nj].q[0] = p[hf];
        kb[nj].q[1] = p[2 + hf];
    }
#pragma unroll
    for (int mi = 0; mi < 4; ++mi)
#pragma unroll
        for (int nj = 0; nj < 4; ++nj)
            s[mi][nj] = wmma_bf16(qa[mi].v, kb[nj].v, s[mi][nj]);

    // scale + shift-mask + softmax (row-wise, rows live in 16-lane halves)
    const float scale = 0.17677669529663687f;   // 32^-0.5
    bf16* Pw = P + wave * 64 * 64;
#pragma unroll
    for (int mi = 0; mi < 4; ++mi)
#pragma unroll
        for (int r = 0; r < 8; ++r) {
            int m = mi * 16 + r + 8 * hf;
            int rm = regid[m];
            float e[4];
            float mx = -1e30f;
#pragma unroll
            for (int nj = 0; nj < 4; ++nj) {
                int ct = nj * 16 + lr;
                float v = s[mi][nj][r] * scale + (regid[ct] == rm ? 0.f : -100.f);
                e[nj] = v;
                mx = fmaxf(mx, v);
            }
#pragma unroll
            for (int off = 1; off < 16; off <<= 1) mx = fmaxf(mx, __shfl_xor(mx, off, 32));
            float sum = 0.f;
#pragma unroll
            for (int nj = 0; nj < 4; ++nj) { e[nj] = __expf(e[nj] - mx); sum += e[nj]; }
#pragma unroll
            for (int off = 1; off < 16; off <<= 1) sum += __shfl_xor(sum, off, 32);
            float inv = 1.f / sum;
#pragma unroll
            for (int nj = 0; nj < 4; ++nj) Pw[m * 64 + nj * 16 + lr] = (bf16)(e[nj] * inv);
        }
    __syncthreads();

    // O = P V  (M=64, N=32, K=64 -> 2 K-steps)
    f32x8 oacc[4][2];
#pragma unroll
    for (int mi = 0; mi < 4; ++mi)
#pragma unroll
        for (int nd = 0; nd < 2; ++nd) oacc[mi][nd] = zero8();

#pragma unroll
    for (int ks = 0; ks < 2; ++ks) {
        Frag pa[4], vb[2];
#pragma unroll
        for (int mi = 0; mi < 4; ++mi) {
            const uint4* p = (const uint4*)(Pw + (mi * 16 + lr) * 64 + ks * 32);
            pa[mi].q[0] = p[hf];
            pa[mi].q[1] = p[2 + hf];
        }
#pragma unroll
        for (int nd = 0; nd < 2; ++nd) {
            const uint4* p = (const uint4*)(vt + (nd * 16 + lr) * 64 + ks * 32);
            vb[nd].q[0] = p[hf];
            vb[nd].q[1] = p[2 + hf];
        }
#pragma unroll
        for (int mi = 0; mi < 4; ++mi)
#pragma unroll
            for (int nd = 0; nd < 2; ++nd)
                oacc[mi][nd] = wmma_bf16(pa[mi].v, vb[nd].v, oacc[mi][nd]);
    }

    // store O -> o[t, head*32 + d]
    bf16* obase = o + (size_t)w * NTOK * DIM + wave * 32;
#pragma unroll
    for (int mi = 0; mi < 4; ++mi)
#pragma unroll
        for (int nd = 0; nd < 2; ++nd)
#pragma unroll
            for (int r = 0; r < 8; ++r) {
                int tok = mi * 16 + r + 8 * hf;
                int d = nd * 16 + lr;
                obase[(size_t)tok * DIM + d] = (bf16)oacc[mi][nd][r];
            }
}

// ---------- depthwise 3x3 + bias + exact GELU ----------
__global__ __launch_bounds__(256) void dwconv_kernel(const bf16* __restrict__ h,
                                                     const float* __restrict__ wgt,
                                                     const float* __restrict__ bias,
                                                     bf16* __restrict__ out) {
    size_t gid = (size_t)blockIdx.x * 256 + threadIdx.x;
    int t  = (int)(gid >> 8);
    int c4 = (int)(gid & 255) * 4;
    int bb = t >> 14, yy = (t >> 7) & 127, xx = t & 127;
    float acc[4] = {bias[c4], bias[c4 + 1], bias[c4 + 2], bias[c4 + 3]};
#pragma unroll
    for (int ky = 0; ky < 3; ++ky) {
        int y2 = yy + ky - 1;
        if ((unsigned)y2 >= 128u) continue;
#pragma unroll
        for (int kx = 0; kx < 3; ++kx) {
            int x2 = xx + kx - 1;
            if ((unsigned)x2 >= 128u) continue;
            bf16x4 pv = *(const bf16x4*)(h + ((size_t)((bb << 14) + y2 * 128 + x2)) * HID + c4);
#pragma unroll
            for (int j = 0; j < 4; ++j)
                acc[j] += (float)pv[j] * wgt[(c4 + j) * 9 + ky * 3 + kx];
        }
    }
    bf16x4 ov;
#pragma unroll
    for (int j = 0; j < 4; ++j) {
        float x = acc[j];
        ov[j] = (bf16)(0.5f * x * (1.f + erff(x * 0.70710678118654752f)));
    }
    *(bf16x4*)(out + (size_t)t * HID + c4) = ov;
}

// ---------- launch ----------
extern "C" void kernel_launch(void* const* d_in, const int* in_sizes, int n_in,
                              void* d_out, int out_size, void* d_ws, size_t ws_size,
                              hipStream_t stream) {
    (void)in_sizes; (void)n_in; (void)out_size; (void)ws_size;
    const float* x      = (const float*)d_in[0];
    const float* g1     = (const float*)d_in[1];
    const float* b1     = (const float*)d_in[2];
    const float* qkv_w  = (const float*)d_in[3];
    const float* proj_w = (const float*)d_in[4];
    const float* proj_b = (const float*)d_in[5];
    const float* g2     = (const float*)d_in[6];
    const float* b2     = (const float*)d_in[7];
    const float* fc1_w  = (const float*)d_in[8];
    const float* fc1_b  = (const float*)d_in[9];
    const float* dw_w   = (const float*)d_in[10];
    const float* dw_b   = (const float*)d_in[11];
    const float* fc2_w  = (const float*)d_in[12];
    const float* fc2_b  = (const float*)d_in[13];
    float* out = (float*)d_out;

    char* ws = (char*)d_ws;
    size_t off = 0;
    bf16* wqkv  = (bf16*)(ws + off); off += (size_t)768 * 256 * 2;
    bf16* wproj = (bf16*)(ws + off); off += (size_t)256 * 256 * 2;
    bf16* wfc1  = (bf16*)(ws + off); off += (size_t)1024 * 256 * 2;
    bf16* wfc2  = (bf16*)(ws + off); off += (size_t)256 * 1024 * 2;
    float* xnF  = (float*)(ws + off); off += (size_t)MTOT * DIM * 4;
    bf16*  xnB  = (bf16*)(ws + off);  off += (size_t)MTOT * DIM * 2;
    bf16*  qkvB = (bf16*)(ws + off);  off += (size_t)MTOT * 768 * 2;
    bf16*  oB   = (bf16*)(ws + off);  off += (size_t)MTOT * DIM * 2;
    float* xsF  = (float*)(ws + off); off += (size_t)MTOT * DIM * 4;
    bf16*  xsnB = (bf16*)(ws + off);  off += (size_t)MTOT * DIM * 2;
    bf16*  hB   = (bf16*)(ws + off);  off += (size_t)MTOT * HID * 2;
    bf16*  gB   = (bf16*)(ws + off);  off += (size_t)MTOT * HID * 2;

    // 1. weights -> bf16
    cvt_kernel<<<(768 * 256) / 256, 256, 0, stream>>>(qkv_w, wqkv, 768 * 256);
    cvt_kernel<<<(256 * 256) / 256, 256, 0, stream>>>(proj_w, wproj, 256 * 256);
    cvt_kernel<<<(1024 * 256) / 256, 256, 0, stream>>>(fc1_w, wfc1, 1024 * 256);
    cvt_kernel<<<(256 * 1024) / 256, 256, 0, stream>>>(fc2_w, wfc2, 256 * 1024);

    // 2. roll + window partition + LN1
    ln_kernel<<<MTOT / 8, 256, 0, stream>>>(x, g1, b1, xnF, xnB, 1);

    // 3. QKV GEMM: [131072,256] x [768,256]^T
    gemm_kernel<0><<<dim3(MTOT / 128, 768 / 256), 256, 0, stream>>>(
        xnB, wqkv, nullptr, nullptr, qkvB, nullptr, 768, 256);

    // 4. shifted-window attention
    int smem = 128 + 8 * 64 * 64 * 2 + 8 * 32 * 64 * 2;  // 98432 B
    (void)hipFuncSetAttribute((const void*)attn_kernel,
                              hipFuncAttributeMaxDynamicSharedMemorySize, smem);
    attn_kernel<<<NWIN, 256, smem, stream>>>(qkvB, oB);

    // 5. proj GEMM + bias + residual(xn) + window-reverse/roll scatter -> xs (f32)
    gemm_kernel<1><<<dim3(MTOT / 128, 1), 256, 0, stream>>>(
        oB, wproj, proj_b, xnF, nullptr, xsF, 256, 256);

    // 6. LN2 -> bf16
    ln_kernel<<<MTOT / 8, 256, 0, stream>>>(xsF, g2, b2, nullptr, xsnB, 0);

    // 7. fc1 GEMM + bias -> h (bf16, NHWC)
    gemm_kernel<2><<<dim3(MTOT / 128, HID / 256), 256, 0, stream>>>(
        xsnB, wfc1, fc1_b, nullptr, hB, nullptr, HID, 256);

    // 8. depthwise 3x3 + bias + GELU
    dwconv_kernel<<<(size_t)MTOT * (HID / 4) / 256, 256, 0, stream>>>(hB, dw_w, dw_b, gB);

    // 9. fc2 GEMM + bias + residual(xs) -> out (f32)
    gemm_kernel<3><<<dim3(MTOT / 128, 1), 256, 0, stream>>>(
        gB, wfc2, fc2_b, xsF, nullptr, out, 256, 1024);
}
